// WindowAttention_11965778887198
// MI455X (gfx1250) — matrix operand order repaired and checked by reading, not compiled, not run on hardware
//
#include <hip/hip_runtime.h>
#include <hip/hip_bf16.h>

// ---------------------------------------------------------------------------
// SwinV2 window attention for gfx1250 (MI455X), wave32 + v_wmma_f32_16x16x32_f16
// ---------------------------------------------------------------------------

#define TDIM   384      // channel dim
#define NTOK   64       // tokens per window
#define NHEADS 12
#define HD     32       // head dim
#define NWIN   64       // distinct masks

typedef _Float16 v16h __attribute__((ext_vector_type(16)));
typedef _Float16 v8h  __attribute__((ext_vector_type(8)));
typedef _Float16 v4h  __attribute__((ext_vector_type(4)));
typedef float    v8f  __attribute__((ext_vector_type(8)));

__device__ __forceinline__ v8f wmma16(v16h a, v16h b, v8f c) {
    return __builtin_amdgcn_wmma_f32_16x16x32_f16(
        /*neg_a=*/false, a, /*neg_b=*/false, b,
        /*c_mod=*/(short)0, c, /*reuse_a=*/false, /*reuse_b=*/false);
}

// A-fragment (16x32, f16) from LDS row-major [m][k], ld = stride halves.
// lane L: m = m0 + (L&15); half = L>>4
// element e -> k = k0 + 16*half + e (contiguous 32B)
__device__ __forceinline__ v16h ldsA(const _Float16* s, int stride, int m0, int k0, int lane) {
    int m = m0 + (lane & 15);
    int hf = lane >> 4;
    const _Float16* p = s + m * stride + k0 + 16 * hf;
    v8h lo = *(const v8h*)p;
    v8h hi = *(const v8h*)(p + 8);
    v16h r;
    for (int i = 0; i < 8; ++i) { r[i] = lo[i]; r[i + 8] = hi[i]; }
    return r;
}

// B-fragment (32x16, f16) where B[k][n] = S[n*stride + k]  (k contiguous per lane)
// lane L: n = n0 + (L&15); element e -> k = k0 + e + 16*(L>>4)
__device__ __forceinline__ v16h ldsB_T(const _Float16* s, int stride, int n0, int k0, int lane) {
    int n = n0 + (lane & 15);
    int hf = lane >> 4;
    const _Float16* p = s + n * stride + k0 + 16 * hf;
    v8h lo = *(const v8h*)p;
    v8h hi = *(const v8h*)(p + 8);
    v16h r;
    for (int i = 0; i < 8; ++i) { r[i] = lo[i]; r[i + 8] = hi[i]; }
    return r;
}

// B-fragment where B[k][n] = S[k*stride + n]  (row-major V matrix; strided per lane)
__device__ __forceinline__ v16h ldsB_rm(const _Float16* s, int stride, int k0, int n0, int lane) {
    int n  = n0 + (lane & 15);
    int kb = k0 + 16 * (lane >> 4);
    v16h r;
    for (int i = 0; i < 16; ++i) r[i] = s[(kb + i) * stride + n];
    return r;
}

// B-fragment from global f16 weight, B[k][n] = W[n*ldw + k]
__device__ __forceinline__ v16h gB_h(const _Float16* W, int ldw, int n0, int k0, int lane) {
    int n = n0 + (lane & 15);
    const _Float16* p = W + (size_t)n * ldw + k0 + 16 * (lane >> 4);
    v8h lo = *(const v8h*)p;
    v8h hi = *(const v8h*)(p + 8);
    v16h r;
    for (int i = 0; i < 8; ++i) { r[i] = lo[i]; r[i + 8] = hi[i]; }
    return r;
}

// D tile (16x16 f32) -> LDS f16 row-major. lane L: n = n0+(L&15); VGPR r -> m = m0 + r + 8*(L>>4)
__device__ __forceinline__ void stD_f16(v8f acc, _Float16* s, int stride, int m0, int n0,
                                        int lane, float scale) {
    int n  = n0 + (lane & 15);
    int mb = m0 + 8 * (lane >> 4);
    for (int r = 0; r < 8; ++r) s[(mb + r) * stride + n] = (_Float16)(acc[r] * scale);
}

// ---------------------------------------------------------------------------
// Kernel 1: CPB MLP -> relative-position-bias table (12 x 64 x 64 f32)
// ---------------------------------------------------------------------------
__global__ void cpb_kernel(const float* __restrict__ rpb, const int* __restrict__ rpb_idx,
                           const float* __restrict__ w1, const float* __restrict__ b1,
                           const float* __restrict__ w2, float* __restrict__ bias) {
    __shared__ float tbl[225 * NHEADS];
    int tid = threadIdx.x;
    for (int r = tid; r < 225; r += 256) {
        float x0 = rpb[r * 2 + 0], x1 = rpb[r * 2 + 1];
        float acc[NHEADS];
        for (int h = 0; h < NHEADS; ++h) acc[h] = 0.f;
        for (int c = 0; c < 512; ++c) {
            float hv = fmaxf(0.f, x0 * w1[c * 2] + x1 * w1[c * 2 + 1] + b1[c]);
            for (int h = 0; h < NHEADS; ++h) acc[h] += w2[h * 512 + c] * hv;
        }
        for (int h = 0; h < NHEADS; ++h) tbl[r * NHEADS + h] = acc[h];
    }
    __syncthreads();
    for (int e = tid; e < NHEADS * NTOK * NTOK; e += 256) {
        int h  = e >> 12;       // / 4096
        int ij = e & 4095;
        float t = tbl[rpb_idx[ij] * NHEADS + h];
        bias[e] = 16.f / (1.f + __expf(-t));
    }
}

// ---------------------------------------------------------------------------
// Kernel 2: f32 -> f16 weight conversion (keeps L2 fragment traffic halved)
// ---------------------------------------------------------------------------
__global__ void cvt_kernel(const float* __restrict__ wqkv, const float* __restrict__ wproj,
                           _Float16* __restrict__ wqkv_h, _Float16* __restrict__ wproj_h) {
    int i = blockIdx.x * 256 + threadIdx.x;
    int stride = gridDim.x * 256;
    for (int k = i; k < 3 * TDIM * TDIM; k += stride) wqkv_h[k] = (_Float16)wqkv[k];
    for (int k = i; k < TDIM * TDIM; k += stride)     wproj_h[k] = (_Float16)wproj[k];
}

// ---------------------------------------------------------------------------
// Kernel 3: fused QKV + attention per window.  192 threads = 6 waves,
// each wave owns heads {w, w+6}.  LDS = 48KB x-tile + 6 * 12KB per-wave qkv.
// ---------------------------------------------------------------------------
__global__ __launch_bounds__(192) void attn_kernel(
        const float* __restrict__ x, const float* __restrict__ mask,
        const float* __restrict__ bias, const _Float16* __restrict__ wqkv,
        _Float16* __restrict__ attn_out) {
    __shared__ _Float16 s_x[NTOK * TDIM];           // 48 KB
    __shared__ _Float16 s_qkv[6][3 * NTOK * HD];    // 6 * 12 KB

    int b = blockIdx.x;
    int tid = threadIdx.x, lane = tid & 31, wv = tid >> 5;
    int lc = lane & 15, hf = lane >> 4;

    // ---- load x tile, f32 -> f16 into LDS ----
    const float4* xg = (const float4*)(x + (size_t)b * NTOK * TDIM);
    for (int i = tid; i < NTOK * TDIM / 4; i += 192) {
        float4 v = xg[i];
        ((v4h*)s_x)[i] = (v4h){(_Float16)v.x, (_Float16)v.y, (_Float16)v.z, (_Float16)v.w};
    }
    __syncthreads();

    const float* maskw = mask + (size_t)(b & (NWIN - 1)) * NTOK * NTOK;
    _Float16* sq = s_qkv[wv];
    _Float16* sk = sq + NTOK * HD;
    _Float16* sv = sk + NTOK * HD;
    _Float16* sp = sq;                 // 64x64 probs overlap q+k (dead by then)

    for (int hi = 0; hi < 2; ++hi) {
        int h = wv + hi * 6;
        const float* biash = bias + (size_t)h * NTOK * NTOK;

        // ---- QKV: (64x384) @ W_t,h^T -> 64x32 each, f16 WMMA, f32 acc ----
        for (int t = 0; t < 3; ++t) {
            _Float16* dst = (t == 0) ? sq : (t == 1) ? sk : sv;
            float scl = (t == 0) ? 0.17677669529663687f : 1.0f;   // 1/sqrt(32) folded into q
            const _Float16* wbase = wqkv + ((size_t)t * TDIM + h * HD) * TDIM;
            for (int nt = 0; nt < 2; ++nt) {
                v16h bfrag[12];
                for (int ks = 0; ks < 12; ++ks)
                    bfrag[ks] = gB_h(wbase, TDIM, nt * 16, ks * 32, lane);
                for (int mt = 0; mt < 4; ++mt) {
                    v8f acc = {};
                    for (int ks = 0; ks < 12; ++ks)
                        acc = wmma16(ldsA(s_x, TDIM, mt * 16, ks * 32, lane), bfrag[ks], acc);
                    stD_f16(acc, dst, HD, mt * 16, nt * 16, lane, scl);
                }
            }
        }

        // ---- preload q (A-frags), k^T and v (B-frags) into registers ----
        v16h aq[4], bk[4], bv[2][2];
        for (int i = 0; i < 4; ++i) aq[i] = ldsA(sq, HD, i * 16, 0, lane);
        for (int i = 0; i < 4; ++i) bk[i] = ldsB_T(sk, HD, i * 16, 0, lane);
        for (int ks = 0; ks < 2; ++ks)
            for (int nt = 0; nt < 2; ++nt)
                bv[ks][nt] = ldsB_rm(sv, HD, ks * 32, nt * 16, lane);

        // ---- logits + bias + mask + row softmax (row-block of 16 at a time) ----
        for (int mt = 0; mt < 4; ++mt) {
            v8f acc[4];
            for (int nt = 0; nt < 4; ++nt) {
                v8f z = {};
                acc[nt] = wmma16(aq[mt], bk[nt], z);
            }
            for (int nt = 0; nt < 4; ++nt)
                for (int r = 0; r < 8; ++r) {
                    int i = mt * 16 + r + 8 * hf;
                    int j = nt * 16 + lc;
                    acc[nt][r] += biash[i * NTOK + j] + maskw[i * NTOK + j];
                }
            // D-layout: per VGPR r, lanes 0-15 hold row r, lanes 16-31 hold row r+8,
            // so xor-shuffles with masks 1/2/4/8 reduce each row independently.
            for (int r = 0; r < 8; ++r) {
                float mx = fmaxf(fmaxf(acc[0][r], acc[1][r]), fmaxf(acc[2][r], acc[3][r]));
                for (int s = 1; s <= 8; s <<= 1) mx = fmaxf(mx, __shfl_xor(mx, s, 32));
                float sum = 0.f;
                for (int nt = 0; nt < 4; ++nt) {
                    float e = __expf(acc[nt][r] - mx);
                    acc[nt][r] = e;
                    sum += e;
                }
                for (int s = 1; s <= 8; s <<= 1) sum += __shfl_xor(sum, s, 32);
                float inv = 1.f / sum;
                int i = mt * 16 + r + 8 * hf;
                for (int nt = 0; nt < 4; ++nt)
                    sp[i * NTOK + nt * 16 + lc] = (_Float16)(acc[nt][r] * inv);
            }
        }

        // ---- out_h = P (64x64) @ V (64x32) ----
        _Float16* og = attn_out + (size_t)b * NTOK * TDIM;
        for (int mt = 0; mt < 4; ++mt)
            for (int nt = 0; nt < 2; ++nt) {
                v8f acc = {};
                for (int ks = 0; ks < 2; ++ks)
                    acc = wmma16(ldsA(sp, NTOK, mt * 16, ks * 32, lane), bv[ks][nt], acc);
                int n  = h * HD + nt * 16 + lc;
                int mb = mt * 16 + 8 * hf;
                for (int r = 0; r < 8; ++r)
                    og[(size_t)(mb + r) * TDIM + n] = (_Float16)acc[r];
            }
    }
}

// ---------------------------------------------------------------------------
// Kernel 4: output projection (64x384)@(384x384)^T + b, 256 threads = 8 waves,
// each wave owns 3 of the 24 N-tiles.
// ---------------------------------------------------------------------------
__global__ __launch_bounds__(256) void proj_kernel(
        const _Float16* __restrict__ attn, const _Float16* __restrict__ wproj,
        const float* __restrict__ bproj, float* __restrict__ out) {
    __shared__ _Float16 s_a[NTOK * TDIM];   // 48 KB
    int b = blockIdx.x;
    int tid = threadIdx.x, lane = tid & 31, wv = tid >> 5;
    int lc = lane & 15, hf = lane >> 4;

    const uint4* ag = (const uint4*)(attn + (size_t)b * NTOK * TDIM);
    uint4* sdst = (uint4*)s_a;
    for (int i = tid; i < NTOK * TDIM * 2 / 16; i += 256) sdst[i] = ag[i];
    __syncthreads();

    float* og = out + (size_t)b * NTOK * TDIM;
    for (int i = 0; i < 3; ++i) {
        int n0 = (wv * 3 + i) * 16;
        v16h bf[12];
        for (int ks = 0; ks < 12; ++ks) bf[ks] = gB_h(wproj, TDIM, n0, ks * 32, lane);
        int n = n0 + lc;
        float bp = bproj[n];
        for (int mt = 0; mt < 4; ++mt) {
            v8f acc = {};
            for (int ks = 0; ks < 12; ++ks)
                acc = wmma16(ldsA(s_a, TDIM, mt * 16, ks * 32, lane), bf[ks], acc);
            int mb = mt * 16 + 8 * hf;
            for (int r = 0; r < 8; ++r)
                og[(size_t)(mb + r) * TDIM + n] = acc[r] + bp;
        }
    }
}

// ---------------------------------------------------------------------------
extern "C" void kernel_launch(void* const* d_in, const int* in_sizes, int n_in,
                              void* d_out, int out_size, void* d_ws, size_t ws_size,
                              hipStream_t stream) {
    (void)in_sizes; (void)n_in; (void)out_size; (void)ws_size;
    const float* x      = (const float*)d_in[0];
    const float* mask   = (const float*)d_in[1];
    const float* rpb    = (const float*)d_in[2];
    const int*   rpbidx = (const int*)d_in[3];
    const float* w_qkv  = (const float*)d_in[4];
    const float* w_cpb1 = (const float*)d_in[5];
    const float* b_cpb1 = (const float*)d_in[6];
    const float* w_cpb2 = (const float*)d_in[7];
    const float* w_proj = (const float*)d_in[8];
    const float* b_proj = (const float*)d_in[9];

    char* ws = (char*)d_ws;
    float*    bias    = (float*)ws;                                   // 196,608 B
    _Float16* wqkv_h  = (_Float16*)(ws + 196608);                     // 884,736 B
    _Float16* wproj_h = (_Float16*)(ws + 196608 + 884736);            // 294,912 B
    _Float16* attn    = (_Float16*)(ws + 196608 + 884736 + 294912);   // 96 MB

    cpb_kernel<<<1, 256, 0, stream>>>(rpb, rpbidx, w_cpb1, b_cpb1, w_cpb2, bias);
    cvt_kernel<<<512, 256, 0, stream>>>(w_qkv, w_proj, wqkv_h, wproj_h);
    attn_kernel<<<2048, 192, 0, stream>>>(x, mask, bias, wqkv_h, attn);
    proj_kernel<<<2048, 256, 0, stream>>>(attn, wproj_h, b_proj, (float*)d_out);
}
